// PTM_25804163514685
// MI455X (gfx1250) — matrix operand order repaired and checked by reading, not compiled, not run on hardware
//
#include <hip/hip_runtime.h>
#include <math.h>

typedef __attribute__((ext_vector_type(16))) __bf16       v16bf;
typedef __attribute__((ext_vector_type(8)))  float        v8f;
typedef __attribute__((ext_vector_type(4)))  float        f32x4;
typedef __attribute__((ext_vector_type(4)))  unsigned int u32x4;

#define B_ 32
#define S_ 8192
#define D_ 256
#define H_ 256

#define ROW_Q   33                    // uint4 per padded LDS row (132 dwords)
#define LDSW_SZ (H_ * ROW_Q * 16)     // 135168 B for W_ctx bf16 (padded)
#define LDS_BYTES (LDSW_SZ + 2 * H_ * 4)

#if __has_builtin(__builtin_amdgcn_tanhf)
#define fast_tanh(x) __builtin_amdgcn_tanhf(x)
#else
#define fast_tanh(x) tanhf(x)
#endif

// ---------------------------------------------------------------------------
// Kernel 1: W_ctx (H x D fp32 row-major) -> bf16 row-major in workspace.
// B[k=d][n=h] = W_ctx[h][d], so row-major [h][d] is per-lane contiguous.
// ---------------------------------------------------------------------------
__global__ __launch_bounds__(256) void wctx_to_bf16_kernel(
    const float* __restrict__ w, __bf16* __restrict__ wbf) {
  int i = blockIdx.x * 256 + threadIdx.x;  // grid covers H_*D_
  wbf[i] = (__bf16)w[i];
}

// ---------------------------------------------------------------------------
// Kernel 2: inp_total[b,h] = sum_k x[b,k]*W_in[h,k] + b_in[h] + b_ctx[h]
// ---------------------------------------------------------------------------
__global__ __launch_bounds__(256) void inp_kernel(
    const float* __restrict__ x, const float* __restrict__ W_in,
    const float* __restrict__ b_in, const float* __restrict__ b_ctx,
    float* __restrict__ inp) {
  int b = blockIdx.x;
  int h = threadIdx.x;
  const float* xr = x + b * H_;
  const float* wr = W_in + h * H_;
  float acc = 0.f;
#pragma unroll 8
  for (int k = 0; k < H_; ++k) acc += xr[k] * wr[k];
  inp[b * H_ + h] = acc + b_in[h] + b_ctx[h];
}

// ---------------------------------------------------------------------------
// Kernel 3: bf16 WMMA GEMM + score epilogue, W_ctx resident in LDS.
// Grid: (S_/512, B_), 256 threads (8 waves). Each wave owns a 16-row M-block;
// H=256 is covered in TWO sequential N-half passes of 8 tiles so only 64
// accumulator VGPRs are live at a time (no scratch spills). Each pass folds
// its tiles into 8 per-row partial scalars before the next pass begins.
// ---------------------------------------------------------------------------
__global__ __launch_bounds__(256) void attn_gemm_kernel(
    const float* __restrict__ ctx, const int* __restrict__ mask,
    const __bf16* __restrict__ wbf, const float* __restrict__ inp,
    const float* __restrict__ V, float* __restrict__ att,
    float* __restrict__ maskout) {
  extern __shared__ char smem[];
  u32x4* lds_w   = (u32x4*)smem;                 // [H_][ROW_Q] padded rows
  float* lds_inp = (float*)(smem + LDSW_SZ);     // [H_]
  float* lds_v   = lds_inp + H_;                 // [H_]

  const int b     = blockIdx.y;
  const int tid   = threadIdx.x;
  const int lane  = tid & 31;
  const int wave  = tid >> 5;                    // = M-block 0..7
  const int khalf = lane >> 4;
  const int nl    = lane & 15;

  // ---- Stage W_ctx bf16 into padded LDS (132-dword rows: 2-way-min banks).
  const u32x4* wq = (const u32x4*)wbf;           // 32 uint4 per 256-bf16 row
  for (int i = tid; i < (H_ * D_) / 8; i += 256) {
    int n = i >> 5, q = i & 31;
    lds_w[n * ROW_Q + q] = wq[i];
  }
  lds_inp[tid] = inp[b * H_ + tid];
  lds_v[tid]   = V[tid];
  __syncthreads();

  const int s_chunk = blockIdx.x * 512;          // 4 iterations of 128 rows

#pragma unroll 1
  for (int it = 0; it < 4; ++it) {
    const int s0 = s_chunk + it * 128 + wave * 16;   // this wave's M-block
    const float* arow = ctx + (size_t)(b * S_ + s0 + nl) * D_;
    if (it < 3) __builtin_prefetch(arow + (size_t)128 * D_, 0, 1);

    float partial[8];
#pragma unroll
    for (int r = 0; r < 8; ++r) partial[r] = 0.f;

#pragma unroll 1
    for (int half = 0; half < 2; ++half) {       // N-half: h in [half*128, +128)
      v8f acc[8] = {};

#pragma unroll
      for (int kk = 0; kk < 8; ++kk) {           // K = 256 in steps of 32
        // A fragment 16x32 bf16: khalf=0 -> K {0..7,16..23}, khalf=1 -> +8.
        const int kb = kk * 32 + khalf * 8;
        f32x4 a0 = *(const f32x4*)(arow + kb);
        f32x4 a1 = *(const f32x4*)(arow + kb + 4);
        f32x4 a2 = *(const f32x4*)(arow + kb + 16);
        f32x4 a3 = *(const f32x4*)(arow + kb + 20);
        v16bf afrag;
#pragma unroll
        for (int j = 0; j < 4; ++j) {
          afrag[j]      = (__bf16)a0[j];
          afrag[4 + j]  = (__bf16)a1[j];
          afrag[8 + j]  = (__bf16)a2[j];
          afrag[12 + j] = (__bf16)a3[j];
        }
#pragma unroll
        for (int nt = 0; nt < 8; ++nt) {
          // B fragment 32x16 bf16 from LDS: lane n = half*128 + nt*16 + nl,
          // lanes 0-15 hold K {0..7,16..23}, lanes 16-31 K +8 (uint4 khalf, khalf+2).
          const u32x4* bp =
              lds_w + (half * 128 + nt * 16 + nl) * ROW_Q + kk * 4 + khalf;
          union { v16bf v; u32x4 q[2]; } bf;
          bf.q[0] = bp[0];
          bf.q[1] = bp[2];
          acc[nt] = __builtin_amdgcn_wmma_f32_16x16x32_bf16(
              false, afrag, false, bf.v, (short)0, acc[nt], false, false);
        }
      }

      // Fold this N-half: lane holds (m = r + khalf*8, h = half*128+nt*16+nl).
#pragma unroll
      for (int r = 0; r < 8; ++r) {
#pragma unroll
        for (int nt = 0; nt < 8; ++nt) {
          const int h = half * 128 + nt * 16 + nl;
          partial[r] += lds_v[h] * fast_tanh(acc[nt][r] + lds_inp[h]);
        }
      }
    }

    // Finish rows: reduce across the 16 N-lanes of each half-wave.
#pragma unroll
    for (int r = 0; r < 8; ++r) {
      float p = partial[r];
      for (int mm = 1; mm < 16; mm <<= 1) p += __shfl_xor(p, mm, 32);
      if (nl == 0) {                      // lanes 0 and 16: full row scores
        const int s  = s0 + khalf * 8 + r;
        const int mk = mask[b * S_ + s];
        const float a = 10.f * fast_tanh(p);
        att[(size_t)b * S_ + s]     = mk ? a : -INFINITY;
        maskout[(size_t)b * S_ + s] = mk ? 1.f : 0.f;
      }
    }
  }
}

// ---------------------------------------------------------------------------
// Kernel 4: per-batch masked argmax + softmax peak: p = 1 / sum(exp(att-max)).
// ---------------------------------------------------------------------------
__global__ __launch_bounds__(256) void finalize_kernel(
    const float* __restrict__ att, float* __restrict__ out) {
  const int b    = blockIdx.x;
  const int tid  = threadIdx.x;
  const int lane = tid & 31;
  const int wave = tid >> 5;
  __shared__ float smax[8];
  __shared__ int   sidx[8];
  __shared__ float ssum[8];
  __shared__ float bmax;
  __shared__ int   bidx_s;

  float best = -INFINITY;
  int   bidx = 0x7fffffff;
  for (int s = tid; s < S_; s += 256) {
    float v = att[(size_t)b * S_ + s];
    if (v > best || (v == best && s < bidx)) { best = v; bidx = s; }
  }
  for (int mm = 16; mm >= 1; mm >>= 1) {
    float ov = __shfl_xor(best, mm, 32);
    int   oi = __shfl_xor(bidx, mm, 32);
    if (ov > best || (ov == best && oi < bidx)) { best = ov; bidx = oi; }
  }
  if (lane == 0) { smax[wave] = best; sidx[wave] = bidx; }
  __syncthreads();
  if (tid == 0) {
    best = smax[0]; bidx = sidx[0];
    for (int w = 1; w < 8; ++w)
      if (smax[w] > best || (smax[w] == best && sidx[w] < bidx)) {
        best = smax[w]; bidx = sidx[w];
      }
    bmax = best; bidx_s = bidx;
  }
  __syncthreads();
  best = bmax;

  float sum = 0.f;
  for (int s = tid; s < S_; s += 256)
    sum += __expf(att[(size_t)b * S_ + s] - best);  // exp(-inf - max) = 0
  for (int mm = 16; mm >= 1; mm >>= 1) sum += __shfl_xor(sum, mm, 32);
  if (lane == 0) ssum[wave] = sum;
  __syncthreads();
  if (tid == 0) {
    float t = 0.f;
    for (int w = 0; w < 8; ++w) t += ssum[w];
    out[b]      = (float)bidx_s;   // argmax index
    out[B_ + b] = 1.f / t;         // max softmax probability
  }
}

// ---------------------------------------------------------------------------
extern "C" void kernel_launch(void* const* d_in, const int* in_sizes, int n_in,
                              void* d_out, int out_size, void* d_ws,
                              size_t ws_size, hipStream_t stream) {
  const float* x       = (const float*)d_in[0];
  const float* context = (const float*)d_in[1];
  const int*   mask    = (const int*)d_in[2];
  const float* W_in    = (const float*)d_in[3];
  const float* b_in    = (const float*)d_in[4];
  const float* W_ctx   = (const float*)d_in[5];
  const float* b_ctx   = (const float*)d_in[6];
  const float* V       = (const float*)d_in[7];
  float*       out     = (float*)d_out;

  char*   ws  = (char*)d_ws;
  __bf16* wbf = (__bf16*)ws;                 // 128 KB: W_ctx bf16
  float*  inp = (float*)(ws + 128 * 1024);   //  32 KB: inp_total[B][H]
  float*  att = (float*)(ws + 160 * 1024);   //   1 MB: att[B][S]

  float* mask_out = out + 2 * B_;            // [indices | p | mask]

  wctx_to_bf16_kernel<<<(H_ * D_) / 256, 256, 0, stream>>>(W_ctx, wbf);
  inp_kernel<<<B_, 256, 0, stream>>>(x, W_in, b_in, b_ctx, inp);
  attn_gemm_kernel<<<dim3(S_ / 512, B_), 256, LDS_BYTES, stream>>>(
      context, mask, wbf, inp, V, att, mask_out);
  finalize_kernel<<<B_, 256, 0, stream>>>(att, out);
}
